// EnhancedMoELayer_64862596104731
// MI455X (gfx1250) — matrix operand order repaired and checked by reading, not compiled, not run on hardware
//
#include <hip/hip_runtime.h>
#include <hip/hip_bf16.h>

typedef __attribute__((ext_vector_type(16))) __bf16 v16bf;
typedef __attribute__((ext_vector_type(8)))  __bf16 v8bf;
typedef __attribute__((ext_vector_type(8)))  float  v8f;

#define NEXP   10      // 8 dense + 2 spike experts
#define D_IN   512
#define HIDDEN 1024
#define D_OUT  256
#define MTILE  64      // tokens per expert block (4 sub-tiles of 16 rows)

// Packed-weight geometry: fragment = 32 rows(K) x 16 cols(N) in WMMA B lane
// order: [e][nt][ks][lane(32)][i(16)] bf16, 1 KB per fragment.
#define W1_NT (HIDDEN / 16)   // 64
#define W1_KS (D_IN / 32)     // 16
#define W2_NT (D_OUT / 16)    // 16
#define W2_KS (HIDDEN / 32)   // 32

// ---------------------------------------------------------------------------
// Kernel 0a/0b: pre-pack W1/W2 fp32 -> bf16 in WMMA-B fragment-major layout.
//   col = lane&15, hi = lane>>4 ; K = ks*32 + hi*16 + i ; N = nt*16 + col
// ---------------------------------------------------------------------------
__global__ __launch_bounds__(256)
void pack_w1_kernel(const float* __restrict__ W1, __bf16* __restrict__ W1p)
{
    const size_t total = (size_t)NEXP * W1_NT * W1_KS * 512;
    const size_t idx = (size_t)blockIdx.x * 256 + threadIdx.x;
    if (idx >= total) return;
    const int i    = idx & 15;
    const int lane = (idx >> 4) & 31;
    const int ks   = (idx >> 9) & (W1_KS - 1);
    const int nt   = (idx >> 13) & (W1_NT - 1);
    const int e    = idx >> 19;
    const int k = ks * 32 + (lane >> 4) * 16 + i;
    const int n = nt * 16 + (lane & 15);
    W1p[idx] = (__bf16)W1[(size_t)e * D_IN * HIDDEN + (size_t)k * HIDDEN + n];
}

__global__ __launch_bounds__(256)
void pack_w2_kernel(const float* __restrict__ W2, __bf16* __restrict__ W2p)
{
    const size_t total = (size_t)NEXP * W2_NT * W2_KS * 512;
    const size_t idx = (size_t)blockIdx.x * 256 + threadIdx.x;
    if (idx >= total) return;
    const int i    = idx & 15;
    const int lane = (idx >> 4) & 31;
    const int ks   = (idx >> 9) & (W2_KS - 1);
    const int nt   = (idx >> 14) & (W2_NT - 1);
    const int e    = idx >> 18;
    const int k = ks * 32 + (lane >> 4) * 16 + i;
    const int n = nt * 16 + (lane & 15);
    W2p[idx] = (__bf16)W2[(size_t)e * HIDDEN * D_OUT + (size_t)k * D_OUT + n];
}

// ---------------------------------------------------------------------------
// Kernel 1: router. logits = x@Wr + br (+ avg_spike on experts 8,9), softmax,
// top-2, normalized gates; scatter (token, gate) into fixed per-expert bins.
// ---------------------------------------------------------------------------
__global__ __launch_bounds__(256)
void moe_router_kernel(const float* __restrict__ x,
                       const float* __restrict__ spike,
                       const float* __restrict__ Wr,
                       const float* __restrict__ br,
                       int B, int T,
                       int*   __restrict__ cursor,   // [NEXP]
                       int*   __restrict__ list,     // [NEXP * B]
                       float* __restrict__ gate)     // [NEXP * B]
{
    __shared__ float sWr[D_IN * NEXP];   // 20 KB
    __shared__ float sBr[NEXP];
    const int tid = threadIdx.x;
    for (int i = tid; i < D_IN * NEXP; i += 256) sWr[i] = Wr[i];
    if (tid < NEXP) sBr[tid] = br[tid];
    __syncthreads();

    const int t = blockIdx.x * 256 + tid;
    if (t >= B) return;

    float acc[NEXP];
#pragma unroll
    for (int e = 0; e < NEXP; ++e) acc[e] = sBr[e];

    const float* xr = x + (size_t)t * D_IN;
    for (int d = 0; d < D_IN; d += 4) {
        const float4 xv = *(const float4*)(xr + d);
#pragma unroll
        for (int e = 0; e < NEXP; ++e) {
            acc[e] += xv.x * sWr[(d + 0) * NEXP + e]
                    + xv.y * sWr[(d + 1) * NEXP + e]
                    + xv.z * sWr[(d + 2) * NEXP + e]
                    + xv.w * sWr[(d + 3) * NEXP + e];
        }
    }

    float s = 0.f;
    const float* sp = spike + (size_t)t * T;
    for (int j = 0; j < T; ++j) s += sp[j];
    s *= (1.0f / (float)T);
    acc[NEXP - 2] += s;
    acc[NEXP - 1] += s;

    float m = acc[0];
#pragma unroll
    for (int e = 1; e < NEXP; ++e) m = fmaxf(m, acc[e]);
    float p[NEXP];
    float sum = 0.f;
#pragma unroll
    for (int e = 0; e < NEXP; ++e) { p[e] = __expf(acc[e] - m); sum += p[e]; }
    const float inv = 1.0f / sum;
#pragma unroll
    for (int e = 0; e < NEXP; ++e) p[e] *= inv;

    int   i0 = 0;  float p0 = p[0];
#pragma unroll
    for (int e = 1; e < NEXP; ++e) if (p[e] > p0) { p0 = p[e]; i0 = e; }
    int   i1 = (i0 == 0) ? 1 : 0;  float p1 = p[i1];
#pragma unroll
    for (int e = 0; e < NEXP; ++e)
        if (e != i0 && p[e] > p1) { p1 = p[e]; i1 = e; }

    const float norm = 1.0f / (p0 + p1 + 1e-9f);
    const float w0 = p0 * norm, w1 = p1 * norm;

    int pos0 = atomicAdd(&cursor[i0], 1);
    list[i0 * B + pos0] = t;  gate[i0 * B + pos0] = w0;
    int pos1 = atomicAdd(&cursor[i1], 1);
    list[i1 * B + pos1] = t;  gate[i1 * B + pos1] = w1;
}

// ---------------------------------------------------------------------------
// Kernel 2: per-expert fused FFN on 64-token tiles using bf16 WMMA with
// pre-packed fragment-major bf16 weights (2x global_load_b128 per B frag).
// ---------------------------------------------------------------------------
__global__ __launch_bounds__(256)
void moe_expert_kernel(const float*  __restrict__ x,
                       const __bf16* __restrict__ W1p,
                       const float*  __restrict__ b1,
                       const __bf16* __restrict__ W2p,
                       const float*  __restrict__ b2,
                       const int*    __restrict__ cursor,
                       const int*    __restrict__ list,
                       const float*  __restrict__ gate,
                       float* __restrict__ out,
                       int B)
{
    __shared__ __bf16 sX[MTILE][D_IN   + 8];   // ~66.5 KB
    __shared__ __bf16 sH[MTILE][HIDDEN + 8];   // ~132 KB
    __shared__ int    sTok[MTILE];
    __shared__ float  sGate[MTILE];

    const int e   = blockIdx.y;
    const int cnt = cursor[e];
    const int t0  = blockIdx.x * MTILE;
    if (t0 >= cnt) return;                  // block-uniform exit

    const int tid = threadIdx.x;
    if (tid < MTILE) {
        const int r = t0 + tid;
        if (r < cnt) { sTok[tid] = list[e * B + r];  sGate[tid] = gate[e * B + r]; }
        else         { sTok[tid] = list[e * B + t0]; sGate[tid] = 0.f; }  // pad row
    }
    __syncthreads();

    // ---- stage X tile as bf16: MTILE rows x 512 cols -----------------------
    for (int idx = tid; idx < MTILE * (D_IN / 4); idx += 256) {
        const int r  = idx >> 7;            // D_IN/4 == 128
        const int c  = (idx & 127) << 2;
        const float4 v = *(const float4*)(x + (size_t)sTok[r] * D_IN + c);
        sX[r][c + 0] = (__bf16)v.x;
        sX[r][c + 1] = (__bf16)v.y;
        sX[r][c + 2] = (__bf16)v.z;
        sX[r][c + 3] = (__bf16)v.w;
    }
    __syncthreads();

    const int wave = tid >> 5;
    const int lane = tid & 31;
    const int col  = lane & 15;
    const int hi   = lane >> 4;

    // ---- GEMM1: H = relu(X @ W1 + b1), H -> LDS as bf16 --------------------
    for (int nt = wave; nt < W1_NT; nt += 8) {
        const int n0 = nt * 16;
        v8f acc[4];
        const float bias = b1[e * HIDDEN + n0 + col];
#pragma unroll
        for (int mm = 0; mm < 4; ++mm)
#pragma unroll
            for (int j = 0; j < 8; ++j) acc[mm][j] = bias;

        const __bf16* wbase =
            W1p + ((size_t)(e * W1_NT + nt) * W1_KS) * 512 + lane * 16;

#pragma unroll 2
        for (int ks = 0; ks < W1_KS; ++ks) {
            // B fragment: 2 contiguous 16B loads per lane (pre-packed bf16)
            const __bf16* wp = wbase + (size_t)ks * 512;
            __builtin_prefetch(wp + 1024, 0, 1);     // 2 fragments ahead
            const v8bf blo = *(const v8bf*)wp;
            const v8bf bhi = *(const v8bf*)(wp + 8);
            const v16bf bm = __builtin_shufflevector(blo, bhi,
                0,1,2,3,4,5,6,7,8,9,10,11,12,13,14,15);

            // hoist all 4 A fragments so the ds_load clause overlaps
            const int kb = ks * 32 + hi * 16;
            v16bf a[4];
#pragma unroll
            for (int mm = 0; mm < 4; ++mm) {
                const int arow = mm * 16 + col;
                const v8bf alo = *(const v8bf*)&sX[arow][kb];
                const v8bf ahi = *(const v8bf*)&sX[arow][kb + 8];
                a[mm] = __builtin_shufflevector(alo, ahi,
                    0,1,2,3,4,5,6,7,8,9,10,11,12,13,14,15);
            }
#pragma unroll
            for (int mm = 0; mm < 4; ++mm)
                acc[mm] = __builtin_amdgcn_wmma_f32_16x16x32_bf16(
                    false, a[mm], false, bm, (short)0, acc[mm], false, false);
        }
#pragma unroll
        for (int mm = 0; mm < 4; ++mm)
#pragma unroll
            for (int j = 0; j < 8; ++j) {
                const int r = mm * 16 + j + hi * 8;
                sH[r][n0 + col] = (__bf16)fmaxf(acc[mm][j], 0.f);
            }
    }
    __syncthreads();

    // ---- GEMM2: Y = H @ W2 + b2; out[t] += gate * Y ------------------------
    for (int nt = wave; nt < W2_NT; nt += 8) {
        const int n0 = nt * 16;
        v8f acc[4];
        const float bias = b2[e * D_OUT + n0 + col];
#pragma unroll
        for (int mm = 0; mm < 4; ++mm)
#pragma unroll
            for (int j = 0; j < 8; ++j) acc[mm][j] = bias;

        const __bf16* wbase =
            W2p + ((size_t)(e * W2_NT + nt) * W2_KS) * 512 + lane * 16;

#pragma unroll 2
        for (int ks = 0; ks < W2_KS; ++ks) {
            const __bf16* wp = wbase + (size_t)ks * 512;
            __builtin_prefetch(wp + 1024, 0, 1);
            const v8bf blo = *(const v8bf*)wp;
            const v8bf bhi = *(const v8bf*)(wp + 8);
            const v16bf bm = __builtin_shufflevector(blo, bhi,
                0,1,2,3,4,5,6,7,8,9,10,11,12,13,14,15);

            const int kb = ks * 32 + hi * 16;
            v16bf a[4];
#pragma unroll
            for (int mm = 0; mm < 4; ++mm) {
                const int arow = mm * 16 + col;
                const v8bf alo = *(const v8bf*)&sH[arow][kb];
                const v8bf ahi = *(const v8bf*)&sH[arow][kb + 8];
                a[mm] = __builtin_shufflevector(alo, ahi,
                    0,1,2,3,4,5,6,7,8,9,10,11,12,13,14,15);
            }
#pragma unroll
            for (int mm = 0; mm < 4; ++mm)
                acc[mm] = __builtin_amdgcn_wmma_f32_16x16x32_bf16(
                    false, a[mm], false, bm, (short)0, acc[mm], false, false);
        }
#pragma unroll
        for (int mm = 0; mm < 4; ++mm)
#pragma unroll
            for (int j = 0; j < 8; ++j) {
                const int r = mm * 16 + j + hi * 8;
                const float w = sGate[r];
                if (w != 0.f)
                    atomicAdd(&out[(size_t)sTok[r] * D_OUT + n0 + col],
                              w * acc[mm][j]);
            }
    }
}

// ---------------------------------------------------------------------------
extern "C" void kernel_launch(void* const* d_in, const int* in_sizes, int n_in,
                              void* d_out, int out_size, void* d_ws, size_t ws_size,
                              hipStream_t stream)
{
    const float* x     = (const float*)d_in[0];
    const float* spike = (const float*)d_in[1];
    const float* Wr    = (const float*)d_in[2];
    const float* br    = (const float*)d_in[3];
    const float* W1    = (const float*)d_in[4];
    const float* b1    = (const float*)d_in[5];
    const float* W2    = (const float*)d_in[6];
    const float* b2    = (const float*)d_in[7];
    float* out = (float*)d_out;

    const int B = in_sizes[0] / D_IN;      // 32768
    const int T = in_sizes[1] / B;         // 16

    // ws layout: [cursor 64B][list][gate][W1p bf16][W2p bf16]
    const size_t n_w1p = (size_t)NEXP * W1_NT * W1_KS * 512;  // 5.24M elems
    const size_t n_w2p = (size_t)NEXP * W2_NT * W2_KS * 512;  // 2.62M elems
    char* w = (char*)d_ws;
    int*    cursor = (int*)w;
    int*    list   = (int*)(w + 64);
    float*  gate   = (float*)(w + 64 + (size_t)NEXP * B * sizeof(int));
    __bf16* W1p    = (__bf16*)(w + 64 + (size_t)NEXP * B * 8);
    __bf16* W2p    = W1p + n_w1p;

    hipMemsetAsync(cursor, 0, 64, stream);
    hipMemsetAsync(d_out, 0, (size_t)out_size * sizeof(float), stream);

    pack_w1_kernel<<<(int)((n_w1p + 255) / 256), 256, 0, stream>>>(W1, W1p);
    pack_w2_kernel<<<(int)((n_w2p + 255) / 256), 256, 0, stream>>>(W2, W2p);

    moe_router_kernel<<<(B + 255) / 256, 256, 0, stream>>>(
        x, spike, Wr, br, B, T, cursor, list, gate);

    dim3 grid((B + MTILE - 1) / MTILE, NEXP);
    moe_expert_kernel<<<grid, 256, 0, stream>>>(
        x, W1p, b1, W2p, b2, cursor, list, gate, out, B);
}
